// SeqMultiHeadAttention_64871186038837
// MI455X (gfx1250) — matrix operand order repaired and checked by reading, not compiled, not run on hardware
//
#include <hip/hip_runtime.h>

typedef __attribute__((ext_vector_type(16))) __bf16 v16bf;
typedef __attribute__((ext_vector_type(8)))  float  v8f;
typedef __attribute__((ext_vector_type(4)))  unsigned int u32x4;
typedef __attribute__((ext_vector_type(8)))  int          i32x8;
typedef __attribute__((ext_vector_type(4)))  int          i32x4;

#define BATCH 2
#define SEQL  2048
#define DIN   1024
#define NH    16
#define EH    64

__device__ __forceinline__ int lane_id() {
  return (int)__builtin_amdgcn_mbcnt_lo(~0u, 0u);   // wave32 lane index
}

__device__ __forceinline__ float lane_xor_f32(float v, int mask) {
  int idx = (lane_id() ^ mask) << 2;                // wraps at 32 lanes
  int r = __builtin_amdgcn_ds_bpermute(idx, __builtin_bit_cast(int, v));
  return __builtin_bit_cast(float, r);
}

// pack two f32 -> 2x bf16 -> one b32 LDS store
__device__ __forceinline__ void store_bf16x2(__bf16* dst, float f0, float f1) {
  union { unsigned int u; __bf16 h[2]; } pk;
  pk.h[0] = (__bf16)f0;
  pk.h[1] = (__bf16)f1;
  *reinterpret_cast<unsigned int*>(dst) = pk.u;
}

// ---------------------------------------------------------------------------
// TDM: async 2-D bf16 tile copy global -> LDS (one descriptor, wave-issued).
// D# layout per CDNA5 ISA (groups 0/1; groups 2-4 zero).
// ---------------------------------------------------------------------------
__device__ __forceinline__ void tdm_load_2d_bf16(
    unsigned int lds_off, const void* gptr,
    unsigned int tensor_d0, unsigned int tensor_d1,
    unsigned int tile_d0,   unsigned int tile_d1,
    unsigned long long stride0_elems)
{
  unsigned long long ga = (unsigned long long)(uintptr_t)gptr;
  u32x4 g0 = { 1u,                                   // count=1, no gather
               lds_off,                               // lds_addr [63:32]
               (unsigned int)ga,                      // global_addr lo
               (unsigned int)((ga >> 32) & 0x1FFFFFFu) | (2u << 30) }; // type=2
  i32x8 g1;
  g1[0] = (int)(1u << 16);                                     // data_size=1 -> 2B
  g1[1] = (int)((tensor_d0 & 0xFFFFu) << 16);                  // tensor_dim0[15:0]
  g1[2] = (int)((tensor_d0 >> 16) | ((tensor_d1 & 0xFFFFu) << 16));
  g1[3] = (int)((tensor_d1 >> 16) | (tile_d0 << 16));          // tile_dim0
  g1[4] = (int)(tile_d1);                                      // tile_dim1, tile_dim2=0
  g1[5] = (int)(unsigned int)(stride0_elems & 0xFFFFFFFFu);    // dim0 stride lo
  g1[6] = (int)((unsigned int)(stride0_elems >> 32) & 0xFFFFu);// dim0 stride hi
  g1[7] = 0;                                                   // dim1 stride = 0
  i32x4 g2 = { 0, 0, 0, 0 };
  i32x4 g3 = { 0, 0, 0, 0 };
  i32x8 g4 = { 0, 0, 0, 0, 0, 0, 0, 0 };
  __builtin_amdgcn_tensor_load_to_lds(g0, g1, g2, g3, g4, 0);
}

// ---------------------------------------------------------------------------
// Kernel 1: Q/K/V projection.  Q,K -> [b][h][s][e] bf16.  V -> TRANSPOSED
// [b][h][e][s] bf16 so the attention kernel's V B-fragments are contiguous.
// ---------------------------------------------------------------------------
__global__ __launch_bounds__(256) void qkv_proj(
    const float* __restrict__ x,
    const float* __restrict__ Wq, const float* __restrict__ Wk,
    const float* __restrict__ Wv,
    __bf16* __restrict__ Qb, __bf16* __restrict__ Kb, __bf16* __restrict__ Vt)
{
  __shared__ __bf16 lx[128][32];   // x tile  [s][k]
  __shared__ __bf16 lw[64][32];    // W tile  [e][k] (transposed)

  const int s0    = blockIdx.x * 128;
  const int h     = blockIdx.y;
  const int b     = blockIdx.z / 3;
  const int which = blockIdx.z % 3;

  const float* W = (which == 0) ? Wq : (which == 1) ? Wk : Wv;
  const float* xb = x + (size_t)b * SEQL * DIN;
  const float* Wh = W + (size_t)h * DIN * EH;

  const int t  = threadIdx.x;
  const int ln = t & 31;
  const int m0 = (t >> 5) * 16;

  v8f acc[4];
  #pragma unroll
  for (int nt = 0; nt < 4; ++nt)
    #pragma unroll
    for (int r = 0; r < 8; ++r) acc[nt][r] = 0.0f;

  for (int k0 = 0; k0 < DIN; k0 += 32) {
    // stage x: 128x32, pair-packed -> b32 LDS stores (lanes take col pairs)
    {
      const int cp  = t & 15;          // column pair 0..15
      const int rw0 = t >> 4;          // row 0..15
      #pragma unroll
      for (int i = 0; i < 8; ++i) {
        const int row = rw0 + i * 16;
        const float* src = &xb[(size_t)(s0 + row) * DIN + k0 + cp * 2];
        store_bf16x2(&lx[row][cp * 2], src[0], src[1]);
      }
    }
    // stage W transposed: lw[e][k], pair-packed along k (coalesced over e)
    {
      const int e   = t & 63;
      const int kp0 = t >> 6;          // k-pair 0..3
      #pragma unroll
      for (int i = 0; i < 4; ++i) {
        const int kp = kp0 + i * 4;    // 0..15
        const float f0 = Wh[(size_t)(k0 + kp * 2)     * EH + e];
        const float f1 = Wh[(size_t)(k0 + kp * 2 + 1) * EH + e];
        store_bf16x2(&lw[e][kp * 2], f0, f1);
      }
    }
    __syncthreads();

    v16bf a;
    {
      const int m  = m0 + (ln & 15);
      const int kb = (ln < 16) ? 0 : 16;
      #pragma unroll
      for (int i = 0; i < 8; ++i) {
        a[i]     = lx[m][kb + i];
        a[8 + i] = lx[m][kb + 8 + i];
      }
    }
    #pragma unroll
    for (int nt = 0; nt < 4; ++nt) {
      v16bf bfr;
      const int n  = nt * 16 + (ln & 15);
      const int kb = (ln < 16) ? 0 : 16;
      #pragma unroll
      for (int i = 0; i < 16; ++i) bfr[i] = lw[n][kb + i];
      acc[nt] = __builtin_amdgcn_wmma_f32_16x16x32_bf16(
          false, a, false, bfr, (short)0, acc[nt], false, false);
    }
    __syncthreads();
  }

  const int nl = ln & 15;
  const int mb = (ln < 16) ? 0 : 8;
  if (which < 2) {
    __bf16* Orow = ((which == 0) ? Qb : Kb) + (size_t)(b * NH + h) * SEQL * EH;
    #pragma unroll
    for (int nt = 0; nt < 4; ++nt)
      #pragma unroll
      for (int r = 0; r < 8; ++r)
        Orow[(size_t)(s0 + m0 + mb + r) * EH + nt * 16 + nl] = (__bf16)acc[nt][r];
  } else {
    __bf16* Ot = Vt + (size_t)(b * NH + h) * EH * SEQL;   // [e][s]
    #pragma unroll
    for (int nt = 0; nt < 4; ++nt)
      #pragma unroll
      for (int r = 0; r < 8; ++r)
        Ot[(size_t)(nt * 16 + nl) * SEQL + (s0 + m0 + mb + r)] = (__bf16)acc[nt][r];
  }
}

// ---------------------------------------------------------------------------
// Kernel 2: causal flash attention.  Block = 64 query rows (4 waves x 16).
// Block-uniform key loop; K and V^T tiles DOUBLE-BUFFERED via TDM:
// issue next block's 2 descriptors, s_wait_tensorcnt(2) -> current buffer
// ready while the prefetch overlaps the 8 WMMAs + softmax of this block.
// ---------------------------------------------------------------------------
__global__ __launch_bounds__(128) void attn(
    const __bf16* __restrict__ Qb, const __bf16* __restrict__ Kb,
    const __bf16* __restrict__ Vt, float* __restrict__ out)
{
  __shared__ __bf16 lk[2][32][64];   // K tiles  [buf][key][e]   (2 x 4 KB)
  __shared__ __bf16 lvt[2][64][32];  // V^T tiles [buf][e][key]  (2 x 4 KB)
  __shared__ float  pP[4][16][32];   // per-wave P relayout      (8 KB)

  const int h  = blockIdx.y;
  const int b  = blockIdx.z;
  const int wv = threadIdx.x >> 5;
  const int ln = threadIdx.x & 31;
  const int qbase = blockIdx.x * 64 + wv * 16;

  const size_t bh = (size_t)(b * NH + h) * SEQL * EH;
  const __bf16* Q  = Qb + bh;
  const __bf16* K  = Kb + bh;
  const __bf16* Vh = Vt + bh;        // [e][s]

  v16bf aq[2];
  {
    const int m  = qbase + (ln & 15);
    const int kb = (ln < 16) ? 0 : 16;
    #pragma unroll
    for (int f = 0; f < 2; ++f)
      #pragma unroll
      for (int i = 0; i < 8; ++i) {
        aq[f][i]     = Q[(size_t)m * EH + f * 32 + kb + i];
        aq[f][8 + i] = Q[(size_t)m * EH + f * 32 + kb + 8 + i];
      }
  }

  float mrow[8], lrow[8];
  #pragma unroll
  for (int r = 0; r < 8; ++r) { mrow[r] = -__builtin_inff(); lrow[r] = 0.0f; }
  v8f o[4];
  #pragma unroll
  for (int nt = 0; nt < 4; ++nt)
    #pragma unroll
    for (int r = 0; r < 8; ++r) o[nt][r] = 0.0f;

  const int   nl    = ln & 15;
  const int   hi    = (ln < 16) ? 0 : 8;
  const float scale = 0.125f;
  const float L2E   = 1.44269504088896f;
  const int   nblocks = blockIdx.x * 2 + 2;   // covers keys <= blockIdx.x*64+63

  // prologue: stage block 0 into buffer 0
  if (wv == 0) {
    tdm_load_2d_bf16((unsigned int)(uintptr_t)&lk[0][0][0], K,
                     EH, SEQL, EH, 32, EH);
    tdm_load_2d_bf16((unsigned int)(uintptr_t)&lvt[0][0][0], Vh,
                     SEQL, EH, 32, EH, SEQL);
  }

  for (int ib = 0; ib < nblocks; ++ib) {
    const int j0  = ib * 32;
    const int cur = ib & 1;
    if (wv == 0) {
      if (ib + 1 < nblocks) {        // prefetch next block into other buffer
        const int nb = (ib + 1) & 1;
        tdm_load_2d_bf16((unsigned int)(uintptr_t)&lk[nb][0][0],
                         K + (size_t)(j0 + 32) * EH,
                         EH, SEQL, EH, 32, EH);
        tdm_load_2d_bf16((unsigned int)(uintptr_t)&lvt[nb][0][0],
                         Vh + (size_t)(j0 + 32),
                         SEQL, EH, 32, EH, SEQL);
        __builtin_amdgcn_s_wait_tensorcnt(2);  // current buffer's 2 loads done
      } else {
        __builtin_amdgcn_s_wait_tensorcnt(0);
      }
    }
    __syncthreads();

    // ---- S = Q K^T : two 16x16 tiles over 32 keys (B frags from LDS, b128)
    v8f sc[2];
    #pragma unroll
    for (int nt = 0; nt < 2; ++nt) {
      const int trow = nt * 16 + nl;
      const int eb   = (ln < 16) ? 0 : 16;
      v16bf bk0, bk1;
      #pragma unroll
      for (int i = 0; i < 16; ++i) {
        bk0[i] = lk[cur][trow][eb + i];
        bk1[i] = lk[cur][trow][32 + eb + i];
      }
      v8f z;
      #pragma unroll
      for (int r = 0; r < 8; ++r) z[r] = 0.0f;
      z = __builtin_amdgcn_wmma_f32_16x16x32_bf16(false, aq[0], false, bk0, (short)0, z, false, false);
      z = __builtin_amdgcn_wmma_f32_16x16x32_bf16(false, aq[1], false, bk1, (short)0, z, false, false);
      sc[nt] = z;
    }

    // ---- scale + causal mask + row max
    float vmax[8];
    #pragma unroll
    for (int r = 0; r < 8; ++r) vmax[r] = -__builtin_inff();
    #pragma unroll
    for (int nt = 0; nt < 2; ++nt) {
      const int keyc = j0 + nt * 16 + nl;
      #pragma unroll
      for (int r = 0; r < 8; ++r) {
        const int qr = qbase + hi + r;
        float v = sc[nt][r] * scale;
        v = (keyc <= qr) ? v : -__builtin_inff();
        sc[nt][r] = v;
        vmax[r] = fmaxf(vmax[r], v);
      }
    }
    #pragma unroll
    for (int msk = 1; msk < 16; msk <<= 1)
      #pragma unroll
      for (int r = 0; r < 8; ++r)
        vmax[r] = fmaxf(vmax[r], lane_xor_f32(vmax[r], msk));

    // ---- online softmax
    float corr[8], rsum[8];
    #pragma unroll
    for (int r = 0; r < 8; ++r) {
      const float mn = fmaxf(mrow[r], vmax[r]);
      corr[r] = exp2f((mrow[r] - mn) * L2E);
      mrow[r] = mn;
      rsum[r] = 0.0f;
    }
    #pragma unroll
    for (int nt = 0; nt < 2; ++nt)
      #pragma unroll
      for (int r = 0; r < 8; ++r) {
        const float p = exp2f((sc[nt][r] - mrow[r]) * L2E);
        rsum[r] += p;
        pP[wv][hi + r][nt * 16 + nl] = p;
      }
    #pragma unroll
    for (int msk = 1; msk < 16; msk <<= 1)
      #pragma unroll
      for (int r = 0; r < 8; ++r)
        rsum[r] += lane_xor_f32(rsum[r], msk);
    #pragma unroll
    for (int r = 0; r < 8; ++r) lrow[r] = lrow[r] * corr[r] + rsum[r];
    #pragma unroll
    for (int nt = 0; nt < 4; ++nt)
      #pragma unroll
      for (int r = 0; r < 8; ++r) o[nt][r] *= corr[r];

    asm volatile("s_wait_dscnt 0" ::: "memory");   // intra-wave LDS RAW

    // ---- P as A-fragment
    v16bf ap;
    {
      const int m  = ln & 15;
      const int kb = (ln < 16) ? 0 : 16;
      #pragma unroll
      for (int i = 0; i < 8; ++i) {
        ap[i]     = (__bf16)pP[wv][m][kb + i];
        ap[8 + i] = (__bf16)pP[wv][m][kb + 8 + i];
      }
    }

    // ---- O += P * V  (V B-frags contiguous from LDS V^T tile)
    #pragma unroll
    for (int nt = 0; nt < 4; ++nt) {
      const int e   = nt * 16 + nl;
      const int kb2 = (ln < 16) ? 0 : 16;
      v16bf bv;
      #pragma unroll
      for (int i = 0; i < 16; ++i)
        bv[i] = lvt[cur][e][kb2 + i];
      o[nt] = __builtin_amdgcn_wmma_f32_16x16x32_bf16(
          false, ap, false, bv, (short)0, o[nt], false, false);
    }
    __syncthreads();   // all waves done with buf[cur] before it is re-staged
  }

  float* orow = out + (size_t)b * SEQL * (NH * EH);
  #pragma unroll
  for (int nt = 0; nt < 4; ++nt)
    #pragma unroll
    for (int r = 0; r < 8; ++r) {
      const int srow = qbase + hi + r;
      orow[(size_t)srow * (NH * EH) + h * EH + nt * 16 + nl] = o[nt][r] / lrow[r];
    }
}

extern "C" void kernel_launch(void* const* d_in, const int* in_sizes, int n_in,
                              void* d_out, int out_size, void* d_ws, size_t ws_size,
                              hipStream_t stream) {
  const float* x  = (const float*)d_in[0];
  const float* Wq = (const float*)d_in[1];
  const float* Wk = (const float*)d_in[2];
  const float* Wv = (const float*)d_in[3];
  float* out = (float*)d_out;

  const size_t per = (size_t)BATCH * NH * SEQL * EH;
  __bf16* Qb = (__bf16*)d_ws;
  __bf16* Kb = Qb + per;
  __bf16* Vt = Kb + per;

  qkv_proj<<<dim3(SEQL / 128, NH, BATCH * 3), 256, 0, stream>>>(x, Wq, Wk, Wv, Qb, Kb, Vt);
  attn<<<dim3(SEQL / 64, NH, BATCH), 128, 0, stream>>>(Qb, Kb, Vt, out);
}
